// GATPooling_58763742544062
// MI455X (gfx1250) — matrix operand order repaired and checked by reading, not compiled, not run on hardware
//
#include <hip/hip_runtime.h>
#include <hip/hip_bf16.h>

typedef __attribute__((ext_vector_type(16))) _Float16 v16h;
typedef __attribute__((ext_vector_type(8)))  _Float16 v8h;
typedef __attribute__((ext_vector_type(8)))  float    v8f;

#define HID   128
#define LDSP  136   // f16 pitch: 272B rows = 17*16B chunks; bank stride 4 mod 64
#define SPLIT 16    // blocks per segment in pooling pass

__device__ __forceinline__ int lower_bound_seg(const int* __restrict__ idx, int E, int v) {
  int lo = 0, hi = E;
  while (lo < hi) { int mid = (lo + hi) >> 1; if (idx[mid] < v) lo = mid + 1; else hi = mid; }
  return lo;
}

// ---------------------------------------------------------------------------
// Kernel 1: scores[e] = relu(x[e,:] @ W1^T + b1) . w2 + b2
// One WG = 256 threads = 8 waves; wave handles 32 rows (two 16-row M tiles).
// GEMM via v_wmma_f32_16x16x32_f16, W1 staged to LDS as f16.
// ---------------------------------------------------------------------------
__global__ __launch_bounds__(256) void gat_scores_wmma(
    const float* __restrict__ x, const float* __restrict__ W1,
    const float* __restrict__ b1, const float* __restrict__ w2,
    const float* __restrict__ b2p, float* __restrict__ scores, int E)
{
  __shared__ _Float16 w1h[HID * LDSP];   // 34,816 B
  const int tid = threadIdx.x;

  // Stage W1 (f32 [n][k] row-major) -> LDS f16 with padded pitch.
  {
    const float4* W4 = (const float4*)W1;
    #pragma unroll
    for (int it = 0; it < (HID * HID / 4) / 256; ++it) {
      const int i = it * 256 + tid;
      const float4 v = W4[i];
      const int n = (i * 4) >> 7;
      const int k = (i * 4) & (HID - 1);
      _Float16* d = &w1h[n * LDSP + k];
      d[0] = (_Float16)v.x; d[1] = (_Float16)v.y;
      d[2] = (_Float16)v.z; d[3] = (_Float16)v.w;
    }
  }
  __syncthreads();

  const int wave = tid >> 5;
  const int lane = tid & 31;
  const int half = lane >> 4;     // lane half (K striping)
  const int ln   = lane & 15;

  const long rowBase = (long)blockIdx.x * 256 + wave * 32;

  float b1v[8], w2v[8];
  #pragma unroll
  for (int t = 0; t < 8; ++t) { b1v[t] = b1[t * 16 + ln]; w2v[t] = w2[t * 16 + ln]; }
  const float bias2 = b2p[0];

  v8f acc[2][8];
  #pragma unroll
  for (int m = 0; m < 2; ++m)
    #pragma unroll
    for (int t = 0; t < 8; ++t)
      #pragma unroll
      for (int i = 0; i < 8; ++i) acc[m][t][i] = 0.0f;

  const float* rp[2];
  #pragma unroll
  for (int m = 0; m < 2; ++m) {
    long r = rowBase + m * 16 + ln;
    if (r >= E) r = E - 1;               // clamp (E % 256 == 0 in practice)
    rp[m] = x + (size_t)r * HID;
  }

  #pragma unroll
  for (int s = 0; s < 4; ++s) {
    const int k0 = s * 32;
    const int ka = k0 + half * 8;        // A frag: K in {ka..ka+7, ka+16..ka+23}
    const int kb = k0 + half * 8;        // B frag: K in {kb..kb+7, kb+16..kb+23}
    v16h a[2];
    #pragma unroll
    for (int m = 0; m < 2; ++m) {
      const float4 f0 = *(const float4*)(rp[m] + ka);
      const float4 f1 = *(const float4*)(rp[m] + ka + 4);
      const float4 f2 = *(const float4*)(rp[m] + ka + 16);
      const float4 f3 = *(const float4*)(rp[m] + ka + 20);
      a[m][0]  = (_Float16)f0.x; a[m][1]  = (_Float16)f0.y;
      a[m][2]  = (_Float16)f0.z; a[m][3]  = (_Float16)f0.w;
      a[m][4]  = (_Float16)f1.x; a[m][5]  = (_Float16)f1.y;
      a[m][6]  = (_Float16)f1.z; a[m][7]  = (_Float16)f1.w;
      a[m][8]  = (_Float16)f2.x; a[m][9]  = (_Float16)f2.y;
      a[m][10] = (_Float16)f2.z; a[m][11] = (_Float16)f2.w;
      a[m][12] = (_Float16)f3.x; a[m][13] = (_Float16)f3.y;
      a[m][14] = (_Float16)f3.z; a[m][15] = (_Float16)f3.w;
    }
    #pragma unroll
    for (int t = 0; t < 8; ++t) {
      const _Float16* bp = &w1h[(t * 16 + ln) * LDSP + kb];  // 16B aligned
      const v8h blo = *(const v8h*)bp;
      const v8h bhi = *(const v8h*)(bp + 16);
      v16h b;
      #pragma unroll
      for (int i = 0; i < 8; ++i) { b[i] = blo[i]; b[8 + i] = bhi[i]; }
      acc[0][t] = __builtin_amdgcn_wmma_f32_16x16x32_f16(
          false, a[0], false, b, (short)0, acc[0][t], false, false);
      acc[1][t] = __builtin_amdgcn_wmma_f32_16x16x32_f16(
          false, a[1], false, b, (short)0, acc[1][t], false, false);
    }
  }

  // Epilogue: relu(+b1) . w2, reduce across the 16-lane N axis, emit scores.
  #pragma unroll
  for (int m = 0; m < 2; ++m) {
    float p[8];
    #pragma unroll
    for (int r = 0; r < 8; ++r) p[r] = 0.0f;
    #pragma unroll
    for (int t = 0; t < 8; ++t) {
      #pragma unroll
      for (int r = 0; r < 8; ++r) {
        float f = acc[m][t][r] + b1v[t];
        f = fmaxf(f, 0.0f);
        p[r] = fmaf(f, w2v[t], p[r]);
      }
    }
    #pragma unroll
    for (int r = 0; r < 8; ++r) {
      float v = p[r];
      v += __shfl_xor(v, 1);
      v += __shfl_xor(v, 2);
      v += __shfl_xor(v, 4);
      v += __shfl_xor(v, 8);
      p[r] = v;
    }
    if (ln == 0) {
      const long base = rowBase + m * 16 + half * 8;  // C rows: +8 for high half
      #pragma unroll
      for (int r = 0; r < 8; ++r)
        if (base + r < E) scores[base + r] = p[r] + bias2;
    }
  }
}

// ---------------------------------------------------------------------------
// Kernel 2: per-segment max and sum(exp(score - max)). One WG per segment.
// ---------------------------------------------------------------------------
__global__ __launch_bounds__(256) void gat_segstats(
    const int* __restrict__ idx, const float* __restrict__ scores,
    float* __restrict__ segMax, float* __restrict__ segDenom, int E)
{
  __shared__ float red[256];
  const int s = blockIdx.x;
  const int start = lower_bound_seg(idx, E, s);
  const int end   = lower_bound_seg(idx, E, s + 1);
  const int tid = threadIdx.x;

  float m = -3.402823466e+38f;
  for (int e = start + tid; e < end; e += 256) m = fmaxf(m, scores[e]);
  red[tid] = m;
  __syncthreads();
  for (int off = 128; off > 0; off >>= 1) {
    if (tid < off) red[tid] = fmaxf(red[tid], red[tid + off]);
    __syncthreads();
  }
  const float mx = red[0];
  __syncthreads();

  float sum = 0.0f;
  for (int e = start + tid; e < end; e += 256) sum += expf(scores[e] - mx);
  red[tid] = sum;
  __syncthreads();
  for (int off = 128; off > 0; off >>= 1) {
    if (tid < off) red[tid] += red[tid + off];
    __syncthreads();
  }
  if (tid == 0) {
    segMax[s]   = (start < end) ? mx     : 0.0f;
    segDenom[s] = (start < end) ? red[0] : 1.0f;
  }
}

// ---------------------------------------------------------------------------
// Kernel 3a: pooled partial sums. SPLIT blocks per segment with fixed edge
// striding (deterministic summation order). float4-vectorized x reads.
// ---------------------------------------------------------------------------
__global__ __launch_bounds__(256) void gat_pool_partial(
    const int* __restrict__ idx, const float* __restrict__ x,
    const float* __restrict__ scores, const float* __restrict__ segMax,
    const float* __restrict__ segDenom, float* __restrict__ partial, int E)
{
  __shared__ float4 red[8][32];
  const int s = blockIdx.x >> 4;          // segment
  const int j = blockIdx.x & (SPLIT - 1); // split index
  const int start = lower_bound_seg(idx, E, s);
  const int end   = lower_bound_seg(idx, E, s + 1);
  const int tid = threadIdx.x;
  const int r = tid >> 5;                 // 8 edge rows per iteration
  const int c = tid & 31;                 // 32 float4 columns

  const float mx = segMax[s];
  const float dn = segDenom[s];
  const float inv = (dn != 0.0f) ? (1.0f / dn) : 0.0f;

  float4 acc = make_float4(0.f, 0.f, 0.f, 0.f);
  for (int e = start + j * 8 + r; e < end; e += SPLIT * 8) {
    const float w = expf(scores[e] - mx) * inv;
    const float4 v = *(const float4*)(x + (size_t)e * HID + c * 4);
    acc.x = fmaf(v.x, w, acc.x);
    acc.y = fmaf(v.y, w, acc.y);
    acc.z = fmaf(v.z, w, acc.z);
    acc.w = fmaf(v.w, w, acc.w);
  }
  red[r][c] = acc;
  __syncthreads();
  if (r == 0) {
    float4 t = red[0][c];
    #pragma unroll
    for (int q = 1; q < 8; ++q) {
      const float4 u = red[q][c];
      t.x += u.x; t.y += u.y; t.z += u.z; t.w += u.w;
    }
    *(float4*)(partial + (size_t)blockIdx.x * HID + c * 4) = t;
  }
}

// ---------------------------------------------------------------------------
// Kernel 3b: reduce SPLIT partials per segment -> d_out (covers all outputs,
// including empty segments -> zeros).
// ---------------------------------------------------------------------------
__global__ __launch_bounds__(256) void gat_pool_reduce(
    const float* __restrict__ partial, float* __restrict__ out, int total)
{
  const int o = blockIdx.x * 256 + threadIdx.x;
  if (o >= total) return;
  const int s = o >> 7;
  const int h = o & (HID - 1);
  float sum = 0.0f;
  #pragma unroll
  for (int j = 0; j < SPLIT; ++j)
    sum += partial[(size_t)(s * SPLIT + j) * HID + h];
  out[o] = sum;
}

extern "C" void kernel_launch(void* const* d_in, const int* in_sizes, int n_in,
                              void* d_out, int out_size, void* d_ws, size_t ws_size,
                              hipStream_t stream) {
  (void)n_in; (void)ws_size;
  const float* x   = (const float*)d_in[0];
  const int*   idx = (const int*)  d_in[1];
  const float* W1  = (const float*)d_in[2];
  const float* b1  = (const float*)d_in[3];
  const float* w2  = (const float*)d_in[4];
  const float* b2  = (const float*)d_in[5];
  float* out = (float*)d_out;

  const int E = in_sizes[1];          // 1,600,000
  const int B = out_size / HID;       // 256 segments

  // Workspace layout (floats), base 16B-aligned; Ea keeps partials 16B-aligned.
  const size_t Ea = ((size_t)E + 3) & ~(size_t)3;
  float* scores   = (float*)d_ws;
  float* segMax   = scores + Ea;
  float* segDenom = segMax + B;
  float* partial  = segDenom + B;     // B * SPLIT * HID floats (~2 MB)

  const int gridScores = (E + 255) / 256;
  gat_scores_wmma<<<gridScores, 256, 0, stream>>>(x, W1, b1, w2, b2, scores, E);
  gat_segstats<<<B, 256, 0, stream>>>(idx, scores, segMax, segDenom, E);
  gat_pool_partial<<<B * SPLIT, 256, 0, stream>>>(idx, x, scores, segMax, segDenom, partial, E);
  gat_pool_reduce<<<(out_size + 255) / 256, 256, 0, stream>>>(partial, out, out_size);
}
